// SparseEncoderV2_53197464929208
// MI455X (gfx1250) — matrix operand order repaired and checked by reading, not compiled, not run on hardware
//
#include <hip/hip_runtime.h>

typedef __attribute__((ext_vector_type(16))) _Float16 v16h;
typedef __attribute__((ext_vector_type(8)))  _Float16 v8h;
typedef __attribute__((ext_vector_type(8)))  float    v8f;

#define BN_EPS      1e-3f
#define LRELU_SLOPE 0.01f

__global__ void k_zero(unsigned* __restrict__ p, long nwords) {
  long i = (long)blockIdx.x * blockDim.x + threadIdx.x;
  if (i < nwords) p[i] = 0u;
}

__global__ void k_scatter(const float* __restrict__ feat,
                          const int* __restrict__ coors,
                          _Float16* __restrict__ dense,
                          float* __restrict__ mask,
                          int nvox, int H, int W, int C) {
  int t = blockIdx.x * blockDim.x + threadIdx.x;
  int i = t / C, c = t % C;
  if (i >= nvox) return;
  int z = coors[i * 4 + 1], y = coors[i * 4 + 2], x = coors[i * 4 + 3];
  long pos = ((long)z * H + y) * W + x;
  dense[pos * C + c] = (_Float16)feat[(long)i * C + c];
  if (c == 0) mask[pos] = 1.0f;
}

// weights (tap, ci, co) fp32 -> (tap, co, ci) f16 : B-fragment = two 16B v8h runs per lane, 16 ci apart
__global__ void k_wprep(const float* __restrict__ w, _Float16* __restrict__ wo,
                        int taps, int Cin, int Cout) {
  int e = blockIdx.x * blockDim.x + threadIdx.x;
  int n = taps * Cin * Cout;
  if (e >= n) return;
  int tap = e / (Cin * Cout);
  int r   = e % (Cin * Cout);
  int ci  = r / Cout, co = r % Cout;
  wo[(long)tap * Cin * Cout + (long)co * Cin + ci] = (_Float16)w[e];
}

// CIN==16: pack tap pairs -> [pair][co][ci0..31]; ci 0-15 = tap 2p, 16-31 = tap 2p+1.
// Phantom tap (>= taps) zero-filled so the K=32 WMMA needs no predication.
__global__ void k_wprep_pair16(const float* __restrict__ w, _Float16* __restrict__ wo,
                               int taps, int Cout) {
  int e = blockIdx.x * blockDim.x + threadIdx.x;
  int npairs = (taps + 1) / 2;
  int n = npairs * Cout * 32;
  if (e >= n) return;
  int p  = e / (Cout * 32);
  int r  = e % (Cout * 32);
  int co = r / 32, j = r % 32;
  int tap = 2 * p + (j >= 16);
  int ci  = j & 15;
  float v = (tap < taps) ? w[((long)tap * 16 + ci) * Cout + co] : 0.0f;
  wo[(long)p * Cout * 32 + (long)co * 32 + j] = (_Float16)v;
}

__global__ void k_bnprep(const float* __restrict__ g, const float* __restrict__ b,
                         const float* __restrict__ m, const float* __restrict__ v,
                         float* __restrict__ scale, float* __restrict__ shift, int C) {
  int c = blockIdx.x * blockDim.x + threadIdx.x;
  if (c >= C) return;
  float s = g[c] * rsqrtf(v[c] + BN_EPS);
  scale[c] = s;
  shift[c] = b[c] - m[c] * s;
}

__global__ void k_maskdown(const float* __restrict__ mi, float* __restrict__ mo,
                           int Din, int Hin, int Win, int Dout, int Hout, int Wout,
                           int kz, int ky, int kx, int sz, int sy, int sx,
                           int pz, int py, int px) {
  int t = blockIdx.x * blockDim.x + threadIdx.x;
  int n = Dout * Hout * Wout;
  if (t >= n) return;
  int x = t % Wout, y = (t / Wout) % Hout, z = t / (Wout * Hout);
  float mx = 0.0f;
  for (int dz = 0; dz < kz; ++dz) {
    int zi = z * sz + dz - pz;
    if ((unsigned)zi >= (unsigned)Din) continue;
    for (int dy = 0; dy < ky; ++dy) {
      int yi = y * sy + dy - py;
      if ((unsigned)yi >= (unsigned)Hin) continue;
      for (int dx = 0; dx < kx; ++dx) {
        int xi = x * sx + dx - px;
        if ((unsigned)xi >= (unsigned)Win) continue;
        float vv = mi[((long)zi * Hin + yi) * Win + xi];
        mx = vv > mx ? vv : mx;
      }
    }
  }
  mo[t] = mx;
}

__device__ __forceinline__ void conv_epilogue(v8f* acc, int NTr, int lane,
    const float* __restrict__ scale, const float* __restrict__ shift,
    const float* __restrict__ mask, _Float16* __restrict__ out,
    long rowbase, int x0, int Wout, int COUT) {
  int m = lane & 15, hi = lane >> 4;
  for (int t = 0; t < NTr; ++t) {
    int co = t * 16 + m;
    float s  = scale[co];
    float sh = shift[co];
#pragma unroll
    for (int r = 0; r < 8; ++r) {
      int xo = x0 + r + hi * 8;
      if (xo < Wout) {
        float v = acc[t][r] * s + sh;
        v = (v >= 0.0f) ? v : LRELU_SLOPE * v;
        v *= mask[rowbase + xo];
        out[(rowbase + xo) * COUT + co] = (_Float16)v;
      }
    }
  }
}

// CIN==16 specialization: two spatial taps fused per K=32 WMMA.
// A: low K-run from tap 2p's position, high K-run from tap 2p+1's.
// B: pre-paired weights, two unguarded 16B v8h loads per lane per N tile.
template <int COUT>
__global__ __launch_bounds__(256)
void k_conv16(const _Float16* __restrict__ in, _Float16* __restrict__ out,
              const float* __restrict__ mask, const _Float16* __restrict__ wfp,
              const float* __restrict__ scale, const float* __restrict__ shift,
              int Din, int Hin, int Win, int Dout, int Hout, int Wout,
              int kz, int ky, int kx, int sz, int sy, int sx,
              int pz, int py, int px, int ntiles, int wtX) {
  constexpr int NT = COUT / 16;
  int gtid = blockIdx.x * 256 + threadIdx.x;
  int wave = gtid >> 5;
  int lane = threadIdx.x & 31;
  if (wave >= ntiles) return;           // whole-wave exit

  int x0 = (wave % wtX) * 16;
  int y  = (wave / wtX) % Hout;
  int z  = wave / (wtX * Hout);
  int m = lane & 15, hi = lane >> 4, base8 = hi * 8;

  int taps   = kz * ky * kx;
  int npairs = (taps + 1) / 2;
  v8f acc[NT] = {};

  for (int p = 0; p < npairs; ++p) {
    v8h afrag[2] = {};
    bool rowv = false;                  // uniform: any tap of this pair z/y-valid?
#pragma unroll
    for (int q = 0; q < 2; ++q) {
      int tap = 2 * p + q;
      if (tap < taps) {
        int dz = tap / (ky * kx);
        int rr = tap % (ky * kx);
        int dy = rr / kx, dx = rr % kx;
        int zi = z * sz + dz - pz;
        int yi = y * sy + dy - py;
        int xi = (x0 + m) * sx + dx - px;
        bool zyok = (unsigned)zi < (unsigned)Din && (unsigned)yi < (unsigned)Hin;
        rowv = rowv || zyok;            // lane-uniform value
        bool ok = zyok && (unsigned)xi < (unsigned)Win;
        if (ok)
          afrag[q] = *(const v8h*)(in + (((long)zi * Hin + yi) * Win + xi) * 16 + base8);
      }
    }
    if (!rowv) continue;                // wave-uniform skip
    v16h a = __builtin_shufflevector(afrag[0], afrag[1], 0, 1, 2, 3, 4, 5, 6, 7,
                                     8, 9, 10, 11, 12, 13, 14, 15);
    const _Float16* wt = wfp + (long)p * (32 * COUT);
#pragma unroll
    for (int t = 0; t < NT; ++t) {
      v16h b = __builtin_shufflevector(*(const v8h*)(wt + (long)(t * 16 + m) * 32 + hi * 8),
                                       *(const v8h*)(wt + (long)(t * 16 + m) * 32 + 16 + hi * 8),
                                       0, 1, 2, 3, 4, 5, 6, 7, 8, 9, 10, 11, 12, 13, 14, 15);  // 2x16B
      acc[t] = __builtin_amdgcn_wmma_f32_16x16x32_f16(
          false, a, false, b, (short)0, acc[t], false, false);
    }
  }

  long rowbase = ((long)z * Hout + y) * Wout;
  conv_epilogue(acc, NT, lane, scale, shift, mask, out, rowbase, x0, Wout, COUT);
}

// General implicit-GEMM conv (CIN >= 32): full K chunks, unpredicated B loads.
template <int CIN, int COUT>
__global__ __launch_bounds__(256)
void k_conv(const _Float16* __restrict__ in, _Float16* __restrict__ out,
            const float* __restrict__ mask, const _Float16* __restrict__ wf,
            const float* __restrict__ scale, const float* __restrict__ shift,
            int Din, int Hin, int Win, int Dout, int Hout, int Wout,
            int kz, int ky, int kx, int sz, int sy, int sx,
            int pz, int py, int px, int ntiles, int wtX) {
  constexpr int NT  = COUT / 16;
  constexpr int KCH = CIN / 32;
  int gtid = blockIdx.x * 256 + threadIdx.x;
  int wave = gtid >> 5;
  int lane = threadIdx.x & 31;
  if (wave >= ntiles) return;

  int x0 = (wave % wtX) * 16;
  int y  = (wave / wtX) % Hout;
  int z  = wave / (wtX * Hout);
  int m = lane & 15, hi = lane >> 4, base8 = hi * 8;

  v8f acc[NT] = {};

  for (int dz = 0; dz < kz; ++dz) {
    int zi = z * sz + dz - pz;
    if ((unsigned)zi >= (unsigned)Din) continue;
    for (int dy = 0; dy < ky; ++dy) {
      int yi = y * sy + dy - py;
      if ((unsigned)yi >= (unsigned)Hin) continue;
      const _Float16* rowp = in + ((long)zi * Hin + yi) * (long)Win * CIN;
      for (int dx = 0; dx < kx; ++dx) {
        int xi = (x0 + m) * sx + dx - px;
        bool ok = (unsigned)xi < (unsigned)Win;
        bool interior = (x0 * sx + dx - px >= 0) &&
                        ((x0 + 15) * sx + dx - px < Win);   // wave-uniform
        const _Float16* src = rowp + (long)xi * CIN;
        int tap = (dz * ky + dy) * kx + dx;
        const _Float16* wt = wf + (long)tap * (CIN * COUT);

#pragma unroll
        for (int kc = 0; kc < KCH; ++kc) {
          v8h alo = {}, ahi = {};
          if (interior) {
            alo = *(const v8h*)(src + kc * 32 + base8);
            ahi = *(const v8h*)(src + kc * 32 + 16 + base8);
          } else if (ok) {
            alo = *(const v8h*)(src + kc * 32 + base8);
            ahi = *(const v8h*)(src + kc * 32 + 16 + base8);
          }
          v16h a = __builtin_shufflevector(alo, ahi, 0, 1, 2, 3, 4, 5, 6, 7,
                                           8, 9, 10, 11, 12, 13, 14, 15);
          int kb = kc * 32 + hi * 8;
#pragma unroll
          for (int t = 0; t < NT; ++t) {
            v16h b = __builtin_shufflevector(*(const v8h*)(wt + (long)(t * 16 + m) * CIN + kb),
                                             *(const v8h*)(wt + (long)(t * 16 + m) * CIN + kb + 16),
                                             0, 1, 2, 3, 4, 5, 6, 7, 8, 9, 10, 11, 12, 13, 14, 15);  // 2x16B
            acc[t] = __builtin_amdgcn_wmma_f32_16x16x32_f16(
                false, a, false, b, (short)0, acc[t], false, false);
          }
        }
      }
    }
  }

  long rowbase = ((long)z * Hout + y) * Wout;
  conv_epilogue(acc, NT, lane, scale, shift, mask, out, rowbase, x0, Wout, COUT);
}

// (z,y,x,c) f16 -> flat fp32 ((c*D+z)*H+y)*W+x
__global__ void k_final(const _Float16* __restrict__ in, float* __restrict__ out,
                        int D, int H, int W, int C) {
  int t = blockIdx.x * blockDim.x + threadIdx.x;
  int n = C * D * H * W;
  if (t >= n) return;
  int x = t % W, y = (t / W) % H, zc = t / (W * H);
  int z = zc % D, c = zc / D;
  out[t] = (float)in[(((long)z * H + y) * W + x) * C + c];
}

struct ConvArgs {
  const _Float16* in; _Float16* out; const float* mask; const _Float16* wf;
  const float* scale; const float* shift;
  int Din, Hin, Win, Dout, Hout, Wout;
  int kz, ky, kx, sz, sy, sx, pz, py, px;
  int ntiles, wtX;
};

template <int COUT>
static void launch_conv16_t(const ConvArgs& a, hipStream_t s) {
  long nthr = (long)a.ntiles * 32;
  k_conv16<COUT><<<(int)((nthr + 255) / 256), 256, 0, s>>>(
      a.in, a.out, a.mask, a.wf, a.scale, a.shift,
      a.Din, a.Hin, a.Win, a.Dout, a.Hout, a.Wout,
      a.kz, a.ky, a.kx, a.sz, a.sy, a.sx, a.pz, a.py, a.px, a.ntiles, a.wtX);
}

template <int CIN, int COUT>
static void launch_conv_t(const ConvArgs& a, hipStream_t s) {
  long nthr = (long)a.ntiles * 32;
  k_conv<CIN, COUT><<<(int)((nthr + 255) / 256), 256, 0, s>>>(
      a.in, a.out, a.mask, a.wf, a.scale, a.shift,
      a.Din, a.Hin, a.Win, a.Dout, a.Hout, a.Wout,
      a.kz, a.ky, a.kx, a.sz, a.sy, a.sx, a.pz, a.py, a.px, a.ntiles, a.wtX);
}

static void launch_conv(int CIN, int COUT, const ConvArgs& a, hipStream_t s) {
  if      (CIN == 16 && COUT == 16) launch_conv16_t<16>(a, s);
  else if (CIN == 16 && COUT == 32) launch_conv16_t<32>(a, s);
  else if (CIN == 32 && COUT == 32) launch_conv_t<32, 32>(a, s);
  else if (CIN == 32 && COUT == 64) launch_conv_t<32, 64>(a, s);
  else                              launch_conv_t<64, 64>(a, s);
}

extern "C" void kernel_launch(void* const* d_in, const int* in_sizes, int n_in,
                              void* d_out, int out_size, void* d_ws, size_t ws_size,
                              hipStream_t stream) {
  const float* feat  = (const float*)d_in[0];
  const int*   coors = (const int*)d_in[1];
  (void)n_in; (void)out_size; (void)ws_size;

  const int NL = 14;
  const int lt [NL] = {0,0,1,0,0,1,0,0,0,1,0,0,0,1};   // 0=subm, 1=down
  const int lci[NL] = {16,16,16,32,32,32,64,64,64,64,64,64,64,64};
  const int lco[NL] = {16,16,32,32,32,64,64,64,64,64,64,64,64,64};
  int kz[NL],ky[NL],kx[NL],sz[NL],sy[NL],sx[NL],pz[NL],py[NL],px[NL];
  for (int l = 0; l < NL; ++l) {
    if (lt[l] == 0)   { kz[l]=ky[l]=kx[l]=3; sz[l]=sy[l]=sx[l]=1; pz[l]=py[l]=px[l]=1; }
    else if (l == 13) { kz[l]=3; ky[l]=1; kx[l]=1; sz[l]=2; sy[l]=1; sx[l]=1; pz[l]=py[l]=px[l]=0; }
    else              { kz[l]=ky[l]=kx[l]=3; sz[l]=sy[l]=sx[l]=2; pz[l]=py[l]=px[l]=1; }
  }
  int Din[NL],Hin[NL],Win[NL],Dout[NL],Hout[NL],Wout[NL];
  int d = 41, h = 128, w = 96;
  for (int l = 0; l < NL; ++l) {
    Din[l]=d; Hin[l]=h; Win[l]=w;
    if (lt[l] == 0) { Dout[l]=d; Hout[l]=h; Wout[l]=w; }
    else {
      Dout[l] = (d + 2*pz[l] - kz[l]) / sz[l] + 1;
      Hout[l] = (h + 2*py[l] - ky[l]) / sy[l] + 1;
      Wout[l] = (w + 2*px[l] - kx[l]) / sx[l] + 1;
    }
    d = Dout[l]; h = Hout[l]; w = Wout[l];
  }

  // deterministic workspace carve
  char* wsp = (char*)d_ws;
  size_t off = 0;
  auto carve = [&](size_t bytes) -> void* {
    void* p = wsp + off;
    off += (bytes + 255) & ~(size_t)255;
    return p;
  };
  const long pos0 = 41L * 128 * 96;
  _Float16* bufA = (_Float16*)carve((size_t)pos0 * 16 * sizeof(_Float16));
  _Float16* bufB = (_Float16*)carve((size_t)pos0 * 16 * sizeof(_Float16));
  float*    mask0 = (float*)carve((size_t)pos0 * sizeof(float));
  float* maskDown[NL] = {};
  for (int l = 0; l < NL; ++l)
    if (lt[l] == 1)
      maskDown[l] = (float*)carve((size_t)Dout[l] * Hout[l] * Wout[l] * sizeof(float));
  _Float16* wf16[NL]; float* lscale[NL]; float* lshift[NL];
  for (int l = 0; l < NL; ++l) {
    long taps = (long)kz[l] * ky[l] * kx[l];
    size_t welems = (lci[l] == 16)
        ? (size_t)((taps + 1) / 2) * 32 * lco[l]   // paired layout
        : (size_t)taps * lci[l] * lco[l];
    wf16[l]   = (_Float16*)carve(welems * sizeof(_Float16));
    lscale[l] = (float*)carve((size_t)lco[l] * sizeof(float));
    lshift[l] = (float*)carve((size_t)lco[l] * sizeof(float));
  }

  // 1) zero dense stage-0 buffer (f16) + mask, scatter voxels
  { long nw = pos0 * 16 * 2 / 4;
    k_zero<<<(int)((nw + 255) / 256), 256, 0, stream>>>((unsigned*)bufA, nw); }
  { long nw = pos0;
    k_zero<<<(int)((nw + 255) / 256), 256, 0, stream>>>((unsigned*)mask0, nw); }
  { int nvox = in_sizes[0] / 16; int n = nvox * 16;
    k_scatter<<<(n + 255) / 256, 256, 0, stream>>>(feat, coors, bufA, mask0, nvox, 128, 96, 16); }

  // 2) weight convert/pack + folded-BN params
  for (int l = 0; l < NL; ++l) {
    const float* W = (const float*)d_in[3 + 5 * l + 0];
    const float* g = (const float*)d_in[3 + 5 * l + 1];
    const float* b = (const float*)d_in[3 + 5 * l + 2];
    const float* m = (const float*)d_in[3 + 5 * l + 3];
    const float* v = (const float*)d_in[3 + 5 * l + 4];
    int taps = kz[l] * ky[l] * kx[l];
    if (lci[l] == 16) {
      int n = ((taps + 1) / 2) * lco[l] * 32;
      k_wprep_pair16<<<(n + 255) / 256, 256, 0, stream>>>(W, wf16[l], taps, lco[l]);
    } else {
      int n = taps * lci[l] * lco[l];
      k_wprep<<<(n + 255) / 256, 256, 0, stream>>>(W, wf16[l], taps, lci[l], lco[l]);
    }
    k_bnprep<<<(lco[l] + 63) / 64, 64, 0, stream>>>(g, b, m, v, lscale[l], lshift[l], lco[l]);
  }

  // 3) layer pipeline (ping-pong f16 buffers)
  _Float16* cur = bufA;
  _Float16* nxt = bufB;
  float* curMask = mask0;
  for (int l = 0; l < NL; ++l) {
    if (lt[l] == 1) {
      int n = Dout[l] * Hout[l] * Wout[l];
      k_maskdown<<<(n + 255) / 256, 256, 0, stream>>>(curMask, maskDown[l],
          Din[l], Hin[l], Win[l], Dout[l], Hout[l], Wout[l],
          kz[l], ky[l], kx[l], sz[l], sy[l], sx[l], pz[l], py[l], px[l]);
      curMask = maskDown[l];
    }
    ConvArgs a;
    a.in = cur; a.out = nxt; a.mask = curMask; a.wf = wf16[l];
    a.scale = lscale[l]; a.shift = lshift[l];
    a.Din = Din[l]; a.Hin = Hin[l]; a.Win = Win[l];
    a.Dout = Dout[l]; a.Hout = Hout[l]; a.Wout = Wout[l];
    a.kz = kz[l]; a.ky = ky[l]; a.kx = kx[l];
    a.sz = sz[l]; a.sy = sy[l]; a.sx = sx[l];
    a.pz = pz[l]; a.py = py[l]; a.px = px[l];
    a.wtX = (Wout[l] + 15) / 16;
    a.ntiles = Dout[l] * Hout[l] * a.wtX;
    launch_conv(lci[l], lco[l], a, stream);
    _Float16* t = cur; cur = nxt; nxt = t;
  }

  // 4) final NDHWC(f16) -> (C*D, H, W) fp32
  { int n = 64 * Dout[13] * Hout[13] * Wout[13];
    k_final<<<(n + 255) / 256, 256, 0, stream>>>(cur, (float*)d_out,
        Dout[13], Hout[13], Wout[13], 64); }
}
